// MCStereo_74071005986973
// MI455X (gfx1250) — matrix operand order repaired and checked by reading, not compiled, not run on hardware
//
#include <hip/hip_runtime.h>
#include <math.h>

#define HT   96
#define WD   160
#define HWSZ (HT*WD)
#define DMX  48
#define KTOP 3
#define RTV  12
#define SV   25   // 2*RT+1

typedef _Float16 half8  __attribute__((ext_vector_type(8)));
typedef _Float16 half16 __attribute__((ext_vector_type(16)));
typedef float    float8 __attribute__((ext_vector_type(8)));

// ---------------------------------------------------------------------------
// Weight pack: fp32 [Cout][Cin][3][3] -> f16 B-fragment-linear blocks.
// Block b = (ot*9 + tap)*nck + ck holds a 32x16 (KxN) tile; lane l supplies
// 16 consecutive halves: N = l&15, K = ck*32 + (l>>4)*16 + t.
// ---------------------------------------------------------------------------
__global__ void pack_w_kernel(const float* __restrict__ w, _Float16* __restrict__ dst,
                              int Cout, int Cin, int nck, int ntiles) {
    int total = ntiles * 9 * nck * 512;
    for (int i = blockIdx.x * blockDim.x + threadIdx.x; i < total;
         i += gridDim.x * blockDim.x) {
        int t   = i & 15;
        int l   = (i >> 4) & 31;
        int b   = i >> 9;
        int ck  = b % nck;
        int tap = (b / nck) % 9;
        int ot  = b / (nck * 9);
        int oc  = ot * 16 + (l & 15);
        int kk  = ck * 32 + ((l >> 4) << 4) + t;
        float v = 0.f;
        if (oc < Cout && kk < Cin) v = w[(oc * Cin + kk) * 9 + tap];
        dst[i] = (_Float16)v;
    }
}

// ---------------------------------------------------------------------------
// Generic 3x3 SAME conv via implicit GEMM + WMMA f16 (fp32 accum).
// Grid: (W/16)*H workgroups, 256 threads (8 waves). Wave w computes output-
// channel tiles {w, w+8} for the 16-pixel row tile of this workgroup.
// WMMA guards are SCALAR (readfirstlane) so EXEC stays all-ones at the WMMA.
// act: 0=none 1=relu 2=sigmoid 3=tanh ; out = act(acc + bias + add) * scale
// ---------------------------------------------------------------------------
__global__ __launch_bounds__(256) void conv3x3_wmma_kernel(
    const float* __restrict__ in, const _Float16* __restrict__ wpack,
    const float* __restrict__ bias, const float* __restrict__ add,
    float* __restrict__ out, int Cin, int Cout, int nck, int ntiles,
    int act, float scale) {
    __shared__ _Float16 Asm[16 * 32];

    const int tid  = threadIdx.x;
    const int lane = tid & 31;
    // Scalarize the wave id so tile guards become s_cbranch (EXEC untouched).
    const int wave = __builtin_amdgcn_readfirstlane(tid >> 5);
    const int tilesPerRow = WD / 16;
    const int y  = blockIdx.x / tilesPerRow;
    const int x0 = (blockIdx.x % tilesPerRow) * 16;

    const int  ot0 = wave;
    const int  ot1 = wave + 8;
    const bool do0 = (ot0 < ntiles);   // scalar
    const bool do1 = (ot1 < ntiles);   // scalar

    float8 acc0 = (float8){0.f, 0.f, 0.f, 0.f, 0.f, 0.f, 0.f, 0.f};
    float8 acc1 = (float8){0.f, 0.f, 0.f, 0.f, 0.f, 0.f, 0.f, 0.f};

    const int aM    = lane & 15;          // matrix row (pixel)
    const int aBase = (lane >> 4) * 16;   // K sub-base per A-fragment layout

    for (int tap = 0; tap < 9; ++tap) {
        const int dy = tap / 3 - 1;
        const int dx = tap % 3 - 1;
        const int yy = y + dy;
        const bool rowOK = (yy >= 0) && (yy < HT);
        for (int ck = 0; ck < nck; ++ck) {
            __syncthreads();
            // Stage 16x32 A tile (fp32 -> f16), zero-padded at borders / c>=Cin.
            for (int e = tid; e < 512; e += 256) {
                int M  = e >> 5;
                int Kk = e & 31;
                int c  = ck * 32 + Kk;
                int xx = x0 + M + dx;
                float v = 0.f;
                if (rowOK && c < Cin && xx >= 0 && xx < WD)
                    v = in[c * HWSZ + yy * WD + xx];
                Asm[M * 32 + Kk] = (_Float16)v;
            }
            __syncthreads();

            // Issue B-fragment global loads first (overlap with DS loads).
            half16 b0, b1;
            if (do0) {
                int blk = (ot0 * 9 + tap) * nck + ck;
                b0 = *(const half16*)&wpack[blk * 512 + lane * 16];
            }
            if (do1) {
                int blk = (ot1 * 9 + tap) * nck + ck;
                b1 = *(const half16*)&wpack[blk * 512 + lane * 16];
            }

            // Per-lane A fragment: elems 0..7 <- K = aBase..aBase+7,
            //                      elems 8..15 <- K = aBase+8..aBase+15
            half16 a;
            {
                half8 lo = *(const half8*)&Asm[aM * 32 + aBase];
                half8 hi = *(const half8*)&Asm[aM * 32 + aBase + 8];
#pragma unroll
                for (int i = 0; i < 8; ++i) { a[i] = lo[i]; a[8 + i] = hi[i]; }
            }
            if (do0)
                acc0 = __builtin_amdgcn_wmma_f32_16x16x32_f16(
                    false, a, false, b0, (short)0, acc0, false, false);
            if (do1)
                acc1 = __builtin_amdgcn_wmma_f32_16x16x32_f16(
                    false, a, false, b1, (short)0, acc1, false, false);
        }
    }

    // Store: D VGPR r -> pixel M = r + 8*(lane>>4), oc = ot*16 + (lane&15)
    if (do0) {
        int oc = ot0 * 16 + (lane & 15);
        if (oc < Cout) {
            float bv = bias[oc];
#pragma unroll
            for (int r = 0; r < 8; ++r) {
                int px  = r + ((lane >> 4) << 3);
                int pix = y * WD + x0 + px;
                float v = acc0[r] + bv;
                if (add) v += add[oc * HWSZ + pix];
                if (act == 1)      v = v > 0.f ? v : 0.f;
                else if (act == 2) v = 1.f / (1.f + expf(-v));
                else if (act == 3) v = tanhf(v);
                out[oc * HWSZ + pix] = v * scale;
            }
        }
    }
    if (do1) {
        int oc = ot1 * 16 + (lane & 15);
        if (oc < Cout) {
            float bv = bias[oc];
#pragma unroll
            for (int r = 0; r < 8; ++r) {
                int px  = r + ((lane >> 4) << 3);
                int pix = y * WD + x0 + px;
                float v = acc1[r] + bv;
                if (add) v += add[oc * HWSZ + pix];
                if (act == 1)      v = v > 0.f ? v : 0.f;
                else if (act == 2) v = 1.f / (1.f + expf(-v));
                else if (act == 3) v = tanhf(v);
                out[oc * HWSZ + pix] = v * scale;
            }
        }
    }
}

// ---------------------------------------------------------------------------
// Correlation volume: vol[d,y,x] = <f1[:,y,x], f2[:,y,x-d]>/16, 0 if x<d
// ---------------------------------------------------------------------------
__global__ void corr_vol_kernel(const float* __restrict__ f1,
                                const float* __restrict__ f2,
                                float* __restrict__ vol) {
    int idx = blockIdx.x * blockDim.x + threadIdx.x;
    if (idx >= DMX * HWSZ) return;
    int d   = idx / HWSZ;
    int pix = idx % HWSZ;
    int x   = pix % WD;
    float s = 0.f;
    if (x >= d) {
        const float* p1 = f1 + pix;
        const float* p2 = f2 + pix - d;
        for (int c = 0; c < 256; ++c) s += p1[c * HWSZ] * p2[c * HWSZ];
        s *= 0.0625f;  // 1/sqrt(256)
    }
    vol[idx] = s;
}

__global__ void pool_vol_kernel(const float* __restrict__ vol,
                                float* __restrict__ vol1) {
    int idx = blockIdx.x * blockDim.x + threadIdx.x;
    if (idx >= (DMX / 2) * HWSZ) return;
    int d   = idx / HWSZ;
    int pix = idx % HWSZ;
    vol1[idx] = 0.5f * (vol[(2 * d) * HWSZ + pix] + vol[(2 * d + 1) * HWSZ + pix]);
}

// top-3 of softmax(vol) == top-3 of vol (monotonic); stable (lower idx first)
__global__ void init_topk_kernel(const float* __restrict__ vol,
                                 float* __restrict__ dt) {
    int pix = blockIdx.x * blockDim.x + threadIdx.x;
    if (pix >= HWSZ) return;
    float v1 = -3.4e38f, v2 = -3.4e38f, v3 = -3.4e38f;
    int i1 = 0, i2 = 0, i3 = 0;
    for (int d = 0; d < DMX; ++d) {
        float p = vol[d * HWSZ + pix];
        if (p > v1)      { v3 = v2; i3 = i2; v2 = v1; i2 = i1; v1 = p; i1 = d; }
        else if (p > v2) { v3 = v2; i3 = i2; v2 = p; i2 = d; }
        else if (p > v3) { v3 = p; i3 = d; }
    }
    dt[0 * HWSZ + pix] = (float)i1;
    dt[1 * HWSZ + pix] = (float)i2;
    dt[2 * HWSZ + pix] = (float)i3;
}

// ---------------------------------------------------------------------------
// Pyramid lookup: build enc_in [153,H,W] (corr 150 + dt 3) and local_cost [75]
// ---------------------------------------------------------------------------
__global__ void lookup_kernel(const float* __restrict__ vol,
                              const float* __restrict__ vol1,
                              const float* __restrict__ dt,
                              float* __restrict__ enc_in,
                              float* __restrict__ lc) {
    int pix = blockIdx.x * blockDim.x + threadIdx.x;
    if (pix >= HWSZ) return;
    for (int k = 0; k < KTOP; ++k) {
        float d0 = dt[k * HWSZ + pix];
        for (int s = 0; s < SV; ++s) {
            // level 0 (D=48, scale 1)
            {
                float pos = d0 + (float)(s - RTV);
                float x0f = floorf(pos);
                float fr  = pos - x0f;
                int xi  = (int)x0f;
                int xi1 = xi + 1;
                float g0 = (xi  >= 0 && xi  < DMX) ? vol[xi  * HWSZ + pix] : 0.f;
                float g1 = (xi1 >= 0 && xi1 < DMX) ? vol[xi1 * HWSZ + pix] : 0.f;
                float v  = g0 * (1.f - fr) + g1 * fr;
                enc_in[(k * 50 + s) * HWSZ + pix] = v;
                lc[(k * SV + s) * HWSZ + pix] = v;
            }
            // level 1 (D=24, scale 2)
            {
                float pos = d0 * 0.5f + (float)(s - RTV);
                float x0f = floorf(pos);
                float fr  = pos - x0f;
                int xi  = (int)x0f;
                int xi1 = xi + 1;
                float g0 = (xi  >= 0 && xi  < DMX / 2) ? vol1[xi  * HWSZ + pix] : 0.f;
                float g1 = (xi1 >= 0 && xi1 < DMX / 2) ? vol1[xi1 * HWSZ + pix] : 0.f;
                enc_in[(k * 50 + SV + s) * HWSZ + pix] = g0 * (1.f - fr) + g1 * fr;
            }
        }
        enc_in[(150 + k) * HWSZ + pix] = d0;
    }
}

// ---------------------------------------------------------------------------
// Pointwise GRU pieces
// ---------------------------------------------------------------------------
__global__ void tanh_init_kernel(const float* __restrict__ net0,
                                 float* __restrict__ hx) {
    int i = blockIdx.x * blockDim.x + threadIdx.x;
    if (i < 128 * HWSZ) hx[i] = tanhf(net0[i]);
}

__global__ void make_qin_kernel(const float* __restrict__ r,
                                const float* __restrict__ hx,
                                float* __restrict__ qin) {
    int i = blockIdx.x * blockDim.x + threadIdx.x;
    if (i >= 128 * HWSZ) return;
    qin[i] = r[i] * hx[i];                       // r * net
    qin[128 * HWSZ + i] = hx[128 * HWSZ + i];    // copy x
}

__global__ void net_update_kernel(const float* __restrict__ z,
                                  const float* __restrict__ q,
                                  float* __restrict__ hx) {
    int i = blockIdx.x * blockDim.x + threadIdx.x;
    if (i >= 128 * HWSZ) return;
    float zz = z[i];
    hx[i] = (1.f - zz) * hx[i] + zz * q[i];
}

// ---------------------------------------------------------------------------
// prob = softmax(local_cost + dlc) over 75; disp = sum(prob*samples);
// disp_topk <- samples at top-3 prob (stable, descending)
// ---------------------------------------------------------------------------
__global__ void prob_disp_kernel(const float* __restrict__ lc,
                                 const float* __restrict__ dlc,
                                 float* __restrict__ dt,
                                 float* __restrict__ disp) {
    int pix = blockIdx.x * blockDim.x + threadIdx.x;
    if (pix >= HWSZ) return;
    float d0[KTOP];
    for (int k = 0; k < KTOP; ++k) d0[k] = dt[k * HWSZ + pix];
    float mx = -3.4e38f;
    for (int ch = 0; ch < KTOP * SV; ++ch) {
        float l = lc[ch * HWSZ + pix] + dlc[ch * HWSZ + pix];
        mx = l > mx ? l : mx;
    }
    float se = 0.f;
    for (int ch = 0; ch < KTOP * SV; ++ch)
        se += expf(lc[ch * HWSZ + pix] + dlc[ch * HWSZ + pix] - mx);
    float inv = 1.f / se;
    float v1 = -1.f, v2 = -1.f, v3 = -1.f;
    int   i1 = 0, i2 = 0, i3 = 0;
    float dsum = 0.f;
    for (int ch = 0; ch < KTOP * SV; ++ch) {
        float p = expf(lc[ch * HWSZ + pix] + dlc[ch * HWSZ + pix] - mx) * inv;
        int k = ch / SV, s = ch % SV;
        float samp = d0[k] + (float)(s - RTV);
        dsum += p * samp;
        if (p > v1)      { v3 = v2; i3 = i2; v2 = v1; i2 = i1; v1 = p; i1 = ch; }
        else if (p > v2) { v3 = v2; i3 = i2; v2 = p; i2 = ch; }
        else if (p > v3) { v3 = p; i3 = ch; }
    }
    disp[pix] = dsum;
    dt[0 * HWSZ + pix] = d0[i1 / SV] + (float)(i1 % SV - RTV);
    dt[1 * HWSZ + pix] = d0[i2 / SV] + (float)(i2 % SV - RTV);
    dt[2 * HWSZ + pix] = d0[i3 / SV] + (float)(i3 % SV - RTV);
}

// ---------------------------------------------------------------------------
// Convex-combination upsample x4 -> [384,640]
// ---------------------------------------------------------------------------
__global__ void upsample_kernel(const float* __restrict__ disp,
                                const float* __restrict__ mask,
                                float* __restrict__ out) {
    int idx = blockIdx.x * blockDim.x + threadIdx.x;
    if (idx >= 4 * HT * 4 * WD) return;
    int X = idx % (4 * WD);
    int Y = idx / (4 * WD);
    int y = Y >> 2, fy = Y & 3, x = X >> 2, fx = X & 3;
    float mv[9];
    float mx = -3.4e38f;
    for (int j = 0; j < 9; ++j) {
        mv[j] = mask[(j * 16 + fy * 4 + fx) * HWSZ + y * WD + x];
        mx = mv[j] > mx ? mv[j] : mx;
    }
    float se = 0.f;
    for (int j = 0; j < 9; ++j) { mv[j] = expf(mv[j] - mx); se += mv[j]; }
    float sum = 0.f;
    for (int j = 0; j < 9; ++j) {
        int di = j / 3 - 1, dj = j % 3 - 1;
        int yy = y + di, xx = x + dj;
        float u = (yy >= 0 && yy < HT && xx >= 0 && xx < WD)
                      ? 4.f * disp[yy * WD + xx] : 0.f;
        sum += mv[j] * u;
    }
    out[idx] = sum / se;
}

// ---------------------------------------------------------------------------
// Host orchestration
// ---------------------------------------------------------------------------
static inline void launch_pack(hipStream_t s, const float* w, _Float16* dst,
                               int Cout, int Cin) {
    int nck = (Cin + 31) / 32, nt = (Cout + 15) / 16;
    int total = nt * 9 * nck * 512;
    pack_w_kernel<<<(total + 255) / 256, 256, 0, s>>>(w, dst, Cout, Cin, nck, nt);
}

static inline void launch_conv(hipStream_t s, const float* in, const _Float16* wp,
                               const float* bias, const float* add, float* out,
                               int Cin, int Cout, int act, float scale) {
    int nck = (Cin + 31) / 32, nt = (Cout + 15) / 16;
    conv3x3_wmma_kernel<<<(WD / 16) * HT, 256, 0, s>>>(in, wp, bias, add, out,
                                                       Cin, Cout, nck, nt, act, scale);
}

extern "C" void kernel_launch(void* const* d_in, const int* in_sizes, int n_in,
                              void* d_out, int out_size, void* d_ws, size_t ws_size,
                              hipStream_t stream) {
    const float* fmap1 = (const float*)d_in[0];
    const float* fmap2 = (const float*)d_in[1];
    const float* net0  = (const float*)d_in[2];
    const float* cz    = (const float*)d_in[3];
    const float* cq    = (const float*)d_in[4];
    const float* cr    = (const float*)d_in[5];
    const float* enc_w = (const float*)d_in[6];
    const float* enc_b = (const float*)d_in[7];
    const float* gz_w  = (const float*)d_in[8];
    const float* gz_b  = (const float*)d_in[9];
    const float* gr_w  = (const float*)d_in[10];
    const float* gr_b  = (const float*)d_in[11];
    const float* gq_w  = (const float*)d_in[12];
    const float* gq_b  = (const float*)d_in[13];
    const float* d1_w  = (const float*)d_in[14];
    const float* d1_b  = (const float*)d_in[15];
    const float* d2_w  = (const float*)d_in[16];
    const float* d2_b  = (const float*)d_in[17];
    const float* m_w   = (const float*)d_in[18];
    const float* m_b   = (const float*)d_in[19];

    char*  ws  = (char*)d_ws;
    size_t off = 0;
    auto alloc = [&](size_t bytes) -> char* {
        char* p = ws + off;
        off += (bytes + 255) & ~(size_t)255;
        return p;
    };
    float* vol    = (float*)alloc(48ull * HWSZ * 4);
    float* vol1   = (float*)alloc(24ull * HWSZ * 4);
    float* dt     = (float*)alloc(3ull * HWSZ * 4);
    float* enc_in = (float*)alloc(153ull * HWSZ * 4);
    float* lc     = (float*)alloc(75ull * HWSZ * 4);
    float* hx     = (float*)alloc(256ull * HWSZ * 4);   // [net | x]
    float* qin    = (float*)alloc(256ull * HWSZ * 4);   // [r*net | x]
    float* zb     = (float*)alloc(128ull * HWSZ * 4);
    float* rb     = (float*)alloc(128ull * HWSZ * 4);
    float* dlc    = (float*)alloc(75ull * HWSZ * 4);
    float* disp   = (float*)alloc((size_t)HWSZ * 4);
    _Float16* wpEnc = (_Float16*)alloc(8ull * 9 * 5 * 512 * 2);
    _Float16* wpZ   = (_Float16*)alloc(8ull * 9 * 8 * 512 * 2);
    _Float16* wpR   = (_Float16*)alloc(8ull * 9 * 8 * 512 * 2);
    _Float16* wpQ   = (_Float16*)alloc(8ull * 9 * 8 * 512 * 2);
    _Float16* wpD1  = (_Float16*)alloc(8ull * 9 * 4 * 512 * 2);
    _Float16* wpD2  = (_Float16*)alloc(5ull * 9 * 4 * 512 * 2);
    _Float16* wpM   = (_Float16*)alloc(9ull * 9 * 4 * 512 * 2);
    // lifetime-safe overlays
    float* qb   = enc_in;  // q written after enc conv consumed enc_in
    float* d1o  = rb;      // d1 out written after r consumed by make_qin
    float* mask = qin;     // mask written after qin consumed by gq conv

    // weight packing (pure function of inputs; deterministic each call)
    launch_pack(stream, enc_w, wpEnc, 128, 153);
    launch_pack(stream, gz_w,  wpZ,   128, 256);
    launch_pack(stream, gr_w,  wpR,   128, 256);
    launch_pack(stream, gq_w,  wpQ,   128, 256);
    launch_pack(stream, d1_w,  wpD1,  128, 128);
    launch_pack(stream, d2_w,  wpD2,  75,  128);
    launch_pack(stream, m_w,   wpM,   144, 128);

    tanh_init_kernel<<<(128 * HWSZ + 255) / 256, 256, 0, stream>>>(net0, hx);
    corr_vol_kernel<<<(DMX * HWSZ + 255) / 256, 256, 0, stream>>>(fmap1, fmap2, vol);
    pool_vol_kernel<<<((DMX / 2) * HWSZ + 255) / 256, 256, 0, stream>>>(vol, vol1);
    init_topk_kernel<<<(HWSZ + 255) / 256, 256, 0, stream>>>(vol, dt);

    int iters = out_size / (4 * HT * 4 * WD);  // 3 for the reference setup
    for (int it = 0; it < iters; ++it) {
        lookup_kernel<<<(HWSZ + 255) / 256, 256, 0, stream>>>(vol, vol1, dt, enc_in, lc);
        // motion encoder: x = relu(conv(enc_in))
        launch_conv(stream, enc_in, wpEnc, enc_b, nullptr, hx + 128 * HWSZ, 153, 128, 1, 1.f);
        // GRU gates
        launch_conv(stream, hx, wpZ, gz_b, cz, zb, 256, 128, 2, 1.f);
        launch_conv(stream, hx, wpR, gr_b, cr, rb, 256, 128, 2, 1.f);
        make_qin_kernel<<<(128 * HWSZ + 255) / 256, 256, 0, stream>>>(rb, hx, qin);
        launch_conv(stream, qin, wpQ, gq_b, cq, qb, 256, 128, 3, 1.f);
        net_update_kernel<<<(128 * HWSZ + 255) / 256, 256, 0, stream>>>(zb, qb, hx);
        // heads
        launch_conv(stream, hx,  wpD1, d1_b, nullptr, d1o, 128, 128, 1, 1.f);
        launch_conv(stream, d1o, wpD2, d2_b, nullptr, dlc, 128, 75, 0, 1.f);
        launch_conv(stream, hx,  wpM,  m_b,  nullptr, mask, 128, 144, 0, 0.25f);
        // joint softmax, soft-argmax, re-topk
        prob_disp_kernel<<<(HWSZ + 255) / 256, 256, 0, stream>>>(lc, dlc, dt, disp);
        // convex upsample into output slice
        upsample_kernel<<<(4 * HT * 4 * WD + 255) / 256, 256, 0, stream>>>(
            disp, mask, (float*)d_out + (size_t)it * 4 * HT * 4 * WD);
    }
}
